// VSAttnLayer_24240795419419
// MI455X (gfx1250) — matrix operand order repaired and checked by reading, not compiled, not run on hardware
//
#include <hip/hip_runtime.h>

typedef __bf16 bf16;
typedef __attribute__((ext_vector_type(16))) __bf16 v16bf;
typedef __attribute__((ext_vector_type(8)))  float  v8f;

constexpr int kC     = 256;
constexpr int kHW    = 96;
constexpr int kHeads = 8;
constexpr int kNq    = 3060;

union FragU { uint4 q[2]; v16bf v; };

__device__ __forceinline__ int laneid() { return (int)(threadIdx.x & 31); }

// ---- WMMA fragment loaders (wave32; contiguous K per lane half, same in A and B, not the ISA table order) ----
// A (16x32 bf16, MxK): lanes 0-15 row=M, elems 0..15 = K(+0..15);
// lanes 16-31 same row, K offset +16.
__device__ __forceinline__ v16bf frag_A(const bf16* __restrict__ base, int ld, int row, int k0) {
  int kb = k0 + ((laneid() >> 4) << 4);
  const bf16* p = base + (size_t)row * ld + kb;
  FragU f;
  f.q[0] = *(const uint4*)(p);
  f.q[1] = *(const uint4*)(p + 8);
  return f.v;
}

// B (32x16 bf16, KxN) with B[k][n] = W[n*ld + k] (nn.Linear-style transposed weight).
// Lanes 0-15: elems 0..15 = K(k0..k0+15); lanes 16-31: K(k0+16..k0+31).
__device__ __forceinline__ v16bf frag_Bt(const bf16* __restrict__ base, int ld, int col, int k0) {
  int kb = k0 + ((laneid() >> 4) << 4);
  const bf16* p = base + (size_t)col * ld + kb;
  FragU f;
  f.q[0] = *(const uint4*)(p);
  f.q[1] = *(const uint4*)(p + 8);
  return f.v;
}

// B (32x16 bf16, KxN) from a row-major KxN source: B[k][n] = src[k*ld + n].
__device__ __forceinline__ v16bf frag_Bn(const bf16* __restrict__ base, int ld, int col, int k0, int kmax) {
  int kb = k0 + ((laneid() >> 4) << 4);
  v16bf v;
#pragma unroll
  for (int i = 0; i < 16; ++i) {
    int k = kb + i; if (k >= kmax) k = kmax - 1;
    v[i] = base[(size_t)k * ld + col];
  }
  return v;
}

// Same loaders but for LDS-resident bf16 panels (lower to ds_load_*).
__device__ __forceinline__ v16bf frag_Bt_lds(const bf16* base, int ld, int col, int k0) {
  int kb = k0 + ((laneid() >> 4) << 4);
  const bf16* p = base + col * ld + kb;
  v16bf v;
#pragma unroll
  for (int i = 0; i < 16; ++i) v[i] = p[i];
  return v;
}

__device__ __forceinline__ v16bf frag_Bn_lds(const bf16* base, int ld, int col, int k0) {
  int kb = k0 + ((laneid() >> 4) << 4);
  v16bf v;
#pragma unroll
  for (int i = 0; i < 16; ++i) v[i] = base[(kb + i) * ld + col];
  return v;
}

// A fragment built from an f32 LDS tile (attention probabilities), converted to bf16.
__device__ __forceinline__ v16bf frag_A_lds(const float* __restrict__ S, int ld, int row, int k0) {
  int kb = k0 + ((laneid() >> 4) << 4);
  const float* p = S + row * ld + kb;
  v16bf v;
#pragma unroll
  for (int i = 0; i < 8; ++i) { v[i] = (bf16)p[i]; v[i + 8] = (bf16)p[i + 8]; }
  return v;
}

__device__ __forceinline__ v8f wmma_bf16(v16bf a, v16bf b, v8f c) {
  return __builtin_amdgcn_wmma_f32_16x16x32_bf16(false, a, false, b, (short)0, c, false, false);
}

// Async global->LDS copy of 16 bytes per lane (ASYNCcnt path, CDNA5 ISA).
__device__ __forceinline__ void async_copy_b128(unsigned lds_addr, const bf16* gaddr) {
  asm volatile("global_load_async_to_lds_b128 %0, %1, off"
               :: "v"(lds_addr), "v"(gaddr) : "memory");
}
__device__ __forceinline__ void async_wait0() {
  asm volatile("s_wait_asynccnt 0x0" ::: "memory");
}

// ---------------------------------------------------------------------------
__global__ void k_f32_to_bf16(const float* __restrict__ in, bf16* __restrict__ out, int n) {
  int i = blockIdx.x * blockDim.x + threadIdx.x;
  if (i < n) out[i] = (bf16)in[i];
}

__global__ void k_zero(float* __restrict__ p, int n) {
  int i = blockIdx.x * blockDim.x + threadIdx.x;
  if (i < n) p[i] = 0.f;
}

// Depthwise 3x3 conv (weight + identity-diagonal kernel) + bias. x,y: (B,256,96,96)
__global__ void k_dwconv(const float* __restrict__ x, const float* __restrict__ w,
                         const float* __restrict__ bias, float* __restrict__ y, int B) {
  int idx = blockIdx.x * blockDim.x + threadIdx.x;
  int total = B * kC * kHW * kHW;
  if (idx >= total) return;
  int xw = idx % kHW; int t = idx / kHW;
  int yh = t % kHW;   t /= kHW;
  int c  = t % kC;    int b = t / kC;
  const float* xp = x + (size_t)(b * kC + c) * kHW * kHW;
  float acc = bias[c];
#pragma unroll
  for (int ky = 0; ky < 3; ++ky)
#pragma unroll
    for (int kx = 0; kx < 3; ++kx) {
      int iy = yh + ky - 1, ix = xw + kx - 1;
      float wv = w[c * 9 + ky * 3 + kx] + ((ky == kx) ? 1.f : 0.f);
      if (iy >= 0 && iy < kHW && ix >= 0 && ix < kHW) acc += wv * xp[iy * kHW + ix];
    }
  y[idx] = acc;
}

// LayerNorm over the channel dim at each (b, spatial) position.
__global__ void k_ln2d(const float* __restrict__ x, const float* __restrict__ g,
                       const float* __restrict__ bta, float* __restrict__ y, int B) {
  int pos = blockIdx.x % (kHW * kHW);
  int b   = blockIdx.x / (kHW * kHW);
  int c   = threadIdx.x;
  size_t idx = (size_t)(b * kC + c) * kHW * kHW + pos;
  float v = x[idx];
  __shared__ float s1[256], s2[256];
  s1[c] = v; s2[c] = v * v; __syncthreads();
  for (int st = 128; st > 0; st >>= 1) {
    if (c < st) { s1[c] += s1[c + st]; s2[c] += s2[c + st]; }
    __syncthreads();
  }
  float mean = s1[0] * (1.f / kC);
  float var  = s2[0] * (1.f / kC) - mean * mean;
  y[idx] = (v - mean) * rsqrtf(var + 1e-5f) * g[c] + bta[c];
}

// Grid pooling: stoken[b][c][cell] = mean over h x h cell of xnorm.
__global__ void k_pool(const float* __restrict__ x, float* __restrict__ st, int B, int h) {
  int hh = kHW / h; int ncell = hh * hh;
  int idx = blockIdx.x * blockDim.x + threadIdx.x;
  int total = B * kC * ncell;
  if (idx >= total) return;
  int cell = idx % ncell; int t = idx / ncell;
  int c = t % kC; int b = t / kC;
  int ci = cell / hh, cj = cell % hh;
  const float* xp = x + (size_t)(b * kC + c) * kHW * kHW;
  float acc = 0.f;
  for (int pi = 0; pi < h; ++pi)
    for (int pj = 0; pj < h; ++pj)
      acc += xp[(ci * h + pi) * kHW + cj * h + pj];
  st[(size_t)(b * kC + c) * ncell + cell] = acc / (float)(h * h);
}

// One soft-assignment iteration: affinity softmax + fold/scatter via f32 atomics.
__global__ void k_affinity(const float* __restrict__ xn, const float* __restrict__ st,
                           float* __restrict__ st_acc, float* __restrict__ asum,
                           int B, int h) {
  int hh = kHW / h; int ncell = hh * hh; int P = h * h;
  int cell = blockIdx.x % ncell; int b = blockIdx.x / ncell;
  int ci = cell / hh, cj = cell % hh;
  int tid = threadIdx.x;
  __shared__ float sf9[kC * 9];
  __shared__ float aff[256 * 9];
  for (int i = tid; i < kC * 9; i += blockDim.x) {
    int c = i / 9, k = i % 9;
    int ti = ci + k / 3 - 1, tj = cj + k % 3 - 1;
    float v = 0.f;
    if (ti >= 0 && ti < hh && tj >= 0 && tj < hh)
      v = st[(size_t)(b * kC + c) * ncell + ti * hh + tj];
    sf9[i] = v;
  }
  __syncthreads();
  const float* xb = xn + (size_t)b * kC * kHW * kHW;
  for (int p = tid; p < P; p += blockDim.x) {
    int pi = p / h, pj = p % h;
    int py = ci * h + pi, px = cj * h + pj;
    float lg[9] = {0.f, 0.f, 0.f, 0.f, 0.f, 0.f, 0.f, 0.f, 0.f};
    for (int c = 0; c < kC; ++c) {
      float pv = xb[(size_t)c * kHW * kHW + py * kHW + px];
#pragma unroll
      for (int k = 0; k < 9; ++k) lg[k] += pv * sf9[c * 9 + k];
    }
    float mx = -1e30f;
#pragma unroll
    for (int k = 0; k < 9; ++k) { lg[k] *= 0.0625f; mx = fmaxf(mx, lg[k]); }
    float s = 0.f;
#pragma unroll
    for (int k = 0; k < 9; ++k) { lg[k] = __expf(lg[k] - mx); s += lg[k]; }
    float inv = 1.f / s;
#pragma unroll
    for (int k = 0; k < 9; ++k) aff[p * 9 + k] = lg[k] * inv;
  }
  __syncthreads();
  int c = tid;  // blockDim == 256 == channels
  float acc[9] = {0.f, 0.f, 0.f, 0.f, 0.f, 0.f, 0.f, 0.f, 0.f};
  for (int p = 0; p < P; ++p) {
    int pi = p / h, pj = p % h;
    float pv = xb[(size_t)c * kHW * kHW + (ci * h + pi) * kHW + cj * h + pj];
#pragma unroll
    for (int k = 0; k < 9; ++k) acc[k] += pv * aff[p * 9 + k];
  }
#pragma unroll
  for (int k = 0; k < 9; ++k) {
    int ti = ci + k / 3 - 1, tj = cj + k % 3 - 1;
    if (ti >= 0 && ti < hh && tj >= 0 && tj < hh)
      atomicAdd(&st_acc[(size_t)(b * kC + c) * ncell + ti * hh + tj], acc[k]);
  }
  if (tid < 9) {
    float s = 0.f;
    for (int p = 0; p < P; ++p) s += aff[p * 9 + tid];
    int ti = ci + tid / 3 - 1, tj = cj + tid % 3 - 1;
    if (ti >= 0 && ti < hh && tj >= 0 && tj < hh)
      atomicAdd(&asum[(size_t)b * ncell + ti * hh + tj], s);
  }
}

// Normalize super-tokens and transpose to (B, N, C) bf16 token matrix.
__global__ void k_tokens(const float* __restrict__ st_acc, const float* __restrict__ asum,
                         bf16* __restrict__ t, int B, int h) {
  int hh = kHW / h; int ncell = hh * hh;
  int idx = blockIdx.x * blockDim.x + threadIdx.x;
  int total = B * ncell * kC;
  if (idx >= total) return;
  int c = idx % kC; int tt = idx / kC;
  int n = tt % ncell; int b = tt / ncell;
  float denom = asum[(size_t)b * ncell + n] + 1e-12f;
  t[(size_t)(b * ncell + n) * kC + c] = (bf16)(st_acc[(size_t)(b * kC + c) * ncell + n] / denom);
}

// Generic C = A @ W^T (+bias, +residual, relu); one wave per 16x16 output tile.
__global__ void k_gemm_bt(const bf16* __restrict__ A, int lda, long long sA,
                          const bf16* __restrict__ W,
                          const float* __restrict__ bias,
                          float* __restrict__ Cf, int ldc, long long sCf,
                          bf16* __restrict__ Cb, int ldcb, long long sCb,
                          const float* __restrict__ Res, int ldr, long long sR,
                          int M, int N, int K, int relu) {
  int b = blockIdx.y;
  int wave = threadIdx.x >> 5;
  int lane = laneid();
  int ntn = (N + 15) >> 4;
  int ntm = (M + 15) >> 4;
  int tile = blockIdx.x * (blockDim.x >> 5) + wave;
  if (tile >= ntm * ntn) return;
  int tm = tile / ntn, tn = tile % ntn;
  const bf16* Ab = A + (size_t)b * sA;
  int row  = tm * 16 + (lane & 15); if (row >= M)  row = M - 1;
  int colW = tn * 16 + (lane & 15); if (colW >= N) colW = N - 1;
  v8f acc = {0.f, 0.f, 0.f, 0.f, 0.f, 0.f, 0.f, 0.f};
  for (int k0 = 0; k0 < K; k0 += 32) {
    if (k0 + 32 < K) {  // global_prefetch_b8 next K-slab
      __builtin_prefetch(Ab + (size_t)row * lda + k0 + 32, 0, 1);
      __builtin_prefetch(W + (size_t)colW * K + k0 + 32, 0, 1);
    }
    v16bf a = frag_A(Ab, lda, row, k0);
    v16bf w = frag_Bt(W, K, colW, k0);
    acc = wmma_bf16(a, w, acc);
  }
  int col = tn * 16 + (lane & 15);
  float bi = (bias && col < N) ? bias[col] : 0.f;
  int rb = (lane >> 4) << 3;
#pragma unroll
  for (int r = 0; r < 8; ++r) {
    int m = tm * 16 + rb + r;
    if (m >= M || col >= N) continue;
    float v = acc[r] + bi;
    if (Res)  v += Res[(size_t)b * sR + (size_t)m * ldr + col];
    if (relu && v < 0.f) v = 0.f;
    if (Cf) Cf[(size_t)b * sCf + (size_t)m * ldc + col] = v;
    if (Cb) Cb[(size_t)b * sCb + (size_t)m * ldcb + col] = (bf16)v;
  }
}

// Stage MHSA (hd=32): one wave per (b, head, 16-query tile); LDS score buffer.
__global__ void k_stage_attn(const bf16* __restrict__ qkv, long long sQKV, int N, int Npad,
                             bf16* __restrict__ o, long long sO, float scale) {
  extern __shared__ float S[];  // 16 * Npad floats
  int qt = blockIdx.x, head = blockIdx.y, b = blockIdx.z;
  int lane = laneid();
  const bf16* qkvb = qkv + (size_t)b * sQKV;
  const bf16* Qb = qkvb + head * 32;
  const bf16* Kb = qkvb + 256 + head * 32;
  const bf16* Vb = qkvb + 512 + head * 32;
  int colL = lane & 15;
  int rb = (lane >> 4) << 3;
  int qrow = qt * 16 + colL; if (qrow >= N) qrow = N - 1;
  v16bf qf = frag_A(Qb, 768, qrow, 0);
  int nt = (N + 15) >> 4;
  for (int kt = 0; kt < nt; ++kt) {
    int krow = kt * 16 + colL; if (krow >= N) krow = N - 1;
    v16bf kf = frag_Bt(Kb, 768, krow, 0);
    v8f s = {0.f, 0.f, 0.f, 0.f, 0.f, 0.f, 0.f, 0.f};
    s = wmma_bf16(qf, kf, s);
#pragma unroll
    for (int r = 0; r < 8; ++r)
      S[(rb + r) * Npad + kt * 16 + colL] = s[r] * scale;
  }
  __syncthreads();
  if (lane < 16) {
    float* row = S + lane * Npad;
    float mx = -1e30f;
    for (int c2 = 0; c2 < N; ++c2) mx = fmaxf(mx, row[c2]);
    float sum = 0.f;
    for (int c2 = 0; c2 < N; ++c2) { float e = __expf(row[c2] - mx); row[c2] = e; sum += e; }
    float inv = 1.f / sum;
    for (int c2 = 0; c2 < N; ++c2) row[c2] *= inv;
    for (int c2 = N; c2 < Npad; ++c2) row[c2] = 0.f;
  }
  __syncthreads();
  v8f a0 = {0.f, 0.f, 0.f, 0.f, 0.f, 0.f, 0.f, 0.f};
  v8f a1 = {0.f, 0.f, 0.f, 0.f, 0.f, 0.f, 0.f, 0.f};
  for (int k0 = 0; k0 < Npad; k0 += 32) {
    v16bf pf = frag_A_lds(S, Npad, colL, k0);
    v16bf v0 = frag_Bn(Vb, 768, colL, k0, N);
    v16bf v1 = frag_Bn(Vb, 768, colL + 16, k0, N);
    a0 = wmma_bf16(pf, v0, a0);
    a1 = wmma_bf16(pf, v1, a1);
  }
  bf16* ob = o + (size_t)b * sO;
#pragma unroll
  for (int r = 0; r < 8; ++r) {
    int tok = qt * 16 + rb + r;
    if (tok < N) {
      ob[(size_t)tok * 256 + head * 32 + colL]      = (bf16)a0[r];
      ob[(size_t)tok * 256 + head * 32 + 16 + colL] = (bf16)a1[r];
    }
  }
}

// Language cross-attention (hd=128, L=32): one wave per (b, head, 16-query tile).
// K/V head panels (32x128 bf16, 8KB each) are staged into LDS with async b128
// copies (ASYNCcnt) and fragments are then built with ds loads.
__global__ void k_cross_attn(const bf16* __restrict__ Q, const bf16* __restrict__ Kb,
                             const bf16* __restrict__ Vb, const int* __restrict__ mask,
                             bf16* __restrict__ O, int M, int L) {
  __shared__ float S[16 * 32];
  __shared__ bf16 Ksh[32 * 128];
  __shared__ bf16 Vsh[32 * 128];
  int qt = blockIdx.x, head = blockIdx.y, b = blockIdx.z;
  int lane = laneid();
  int colL = lane & 15;
  int rb = (lane >> 4) << 3;
  const bf16* Qh = Q  + (size_t)b * M * 1024 + head * 128;
  const bf16* Kh = Kb + (size_t)b * L * 1024 + head * 128;
  const bf16* Vh = Vb + (size_t)b * L * 1024 + head * 128;

  // Stage K/V panels: 16 bytes (8 bf16) per lane per issue, 16 issues per panel.
#pragma unroll
  for (int i = 0; i < 16; ++i) {
    int e = (i * 32 + lane) * 8;          // bf16 element offset within the panel
    int key = e >> 7, d = e & 127;
    async_copy_b128((unsigned)(size_t)&Ksh[e], Kh + (size_t)key * 1024 + d);
    async_copy_b128((unsigned)(size_t)&Vsh[e], Vh + (size_t)key * 1024 + d);
  }
  async_wait0();
  __syncthreads();

  int qrow = qt * 16 + colL; if (qrow >= M) qrow = M - 1;
  const float scale = 0.08838834764831845f;  // 1/sqrt(128)
  for (int kt = 0; kt < 2; ++kt) {
    v8f s = {0.f, 0.f, 0.f, 0.f, 0.f, 0.f, 0.f, 0.f};
    for (int kc = 0; kc < 128; kc += 32) {
      v16bf qf = frag_A(Qh, 1024, qrow, kc);
      v16bf kf = frag_Bt_lds(Ksh, 128, kt * 16 + colL, kc);
      s = wmma_bf16(qf, kf, s);
    }
#pragma unroll
    for (int r = 0; r < 8; ++r)
      S[(rb + r) * 32 + kt * 16 + colL] = s[r] * scale;
  }
  __syncthreads();
  if (lane < 16) {
    float* row = S + lane * 32;
    float mx = -1e30f;
    for (int c2 = 0; c2 < 32; ++c2) {
      float lg = fminf(fmaxf(row[c2], -50000.f), 50000.f);
      if (!(mask[b * 32 + c2] > 0)) lg = -1e9f;
      row[c2] = lg; mx = fmaxf(mx, lg);
    }
    float sum = 0.f;
    for (int c2 = 0; c2 < 32; ++c2) { float e = __expf(row[c2] - mx); row[c2] = e; sum += e; }
    float inv = 1.f / sum;
    for (int c2 = 0; c2 < 32; ++c2) row[c2] *= inv;
  }
  __syncthreads();
  v16bf pf = frag_A_lds(S, 32, colL, 0);
  bf16* Oh = O + (size_t)b * M * 1024 + head * 128;
  for (int dt = 0; dt < 8; ++dt) {
    v16bf vf = frag_Bn_lds(Vsh, 128, dt * 16 + colL, 0);
    v8f a = {0.f, 0.f, 0.f, 0.f, 0.f, 0.f, 0.f, 0.f};
    a = wmma_bf16(pf, vf, a);
#pragma unroll
    for (int r = 0; r < 8; ++r) {
      int tok = qt * 16 + rb + r;
      if (tok < M) Oh[(size_t)tok * 1024 + dt * 16 + colL] = (bf16)a[r];
    }
  }
}

// LayerNorm over last dim (256) on (T, 256) rows; optional bf16 shadow copy.
__global__ void k_ln(const float* __restrict__ x, const float* __restrict__ g,
                     const float* __restrict__ bta, float* __restrict__ yf,
                     bf16* __restrict__ yb, int T) {
  int t = blockIdx.x; if (t >= T) return;
  int c = threadIdx.x;
  float v = x[(size_t)t * kC + c];
  __shared__ float s1[256], s2[256];
  s1[c] = v; s2[c] = v * v; __syncthreads();
  for (int st = 128; st > 0; st >>= 1) {
    if (c < st) { s1[c] += s1[c + st]; s2[c] += s2[c + st]; }
    __syncthreads();
  }
  float mean = s1[0] * (1.f / kC);
  float var  = s2[0] * (1.f / kC) - mean * mean;
  float o = (v - mean) * rsqrtf(var + 1e-5f) * g[c] + bta[c];
  if (yf) yf[(size_t)t * kC + c] = o;
  if (yb) yb[(size_t)t * kC + c] = (bf16)o;
}

// ---------------------------------------------------------------------------
extern "C" void kernel_launch(void* const* d_in, const int* in_sizes, int n_in,
                              void* d_out, int out_size, void* d_ws, size_t ws_size,
                              hipStream_t stream) {
  (void)in_sizes; (void)n_in; (void)out_size; (void)ws_size;
  const float* sp[4] = {(const float*)d_in[0], (const float*)d_in[1],
                        (const float*)d_in[2], (const float*)d_in[3]};
  const float* lang   = (const float*)d_in[4];
  const int*   lmask  = (const int*)d_in[5];
  const float* pose_w = (const float*)d_in[6];
  const float* pose_b = (const float*)d_in[7];
  const float* pn_g   = (const float*)d_in[8];
  const float* pn_b   = (const float*)d_in[9];
  const float* qkv_w  = (const float*)d_in[10];
  const float* qkv_b  = (const float*)d_in[11];
  const float* proj_w = (const float*)d_in[12];
  const float* proj_b = (const float*)d_in[13];
  const float* wq = (const float*)d_in[14]; const float* bq = (const float*)d_in[15];
  const float* wk = (const float*)d_in[16]; const float* bk = (const float*)d_in[17];
  const float* wv = (const float*)d_in[18]; const float* bv = (const float*)d_in[19];
  const float* wo = (const float*)d_in[20]; const float* bo = (const float*)d_in[21];
  const float* n1_g = (const float*)d_in[22]; const float* n1_b = (const float*)d_in[23];
  const float* n2_g = (const float*)d_in[24]; const float* n2_b = (const float*)d_in[25];
  const float* m1_w = (const float*)d_in[26]; const float* m1_b = (const float*)d_in[27];
  const float* m2_w = (const float*)d_in[28]; const float* m2_b = (const float*)d_in[29];

  char* ws = (char*)d_ws;
  size_t off = 0;
  auto alloc = [&](size_t bytes) -> void* {
    void* p = ws + off;
    off = (off + bytes + 255) & ~(size_t)255;
    return p;
  };

  const int B = 2;
  bf16* qkvw_bf = (bf16*)alloc((size_t)4 * 768 * 256 * 2);
  bf16* projw_bf = (bf16*)alloc((size_t)4 * 256 * 256 * 2);
  bf16* wq_bf = (bf16*)alloc((size_t)1024 * 256 * 2);
  bf16* wk_bf = (bf16*)alloc((size_t)1024 * 256 * 2);
  bf16* wv_bf = (bf16*)alloc((size_t)1024 * 256 * 2);
  bf16* wo_bf = (bf16*)alloc((size_t)256 * 1024 * 2);
  bf16* m1_bf = (bf16*)alloc((size_t)1024 * 256 * 2);
  bf16* m2_bf = (bf16*)alloc((size_t)256 * 1024 * 2);
  bf16* lang_bf = (bf16*)alloc((size_t)B * 32 * 256 * 2);
  float* xconv = (float*)alloc((size_t)B * kC * kHW * kHW * 4);   // also reused as Q
  float* xnorm = (float*)alloc((size_t)B * kC * kHW * kHW * 4);   // also reused as O / H
  float* stok  = (float*)alloc((size_t)B * kC * 2304 * 4);
  float* stacc = (float*)alloc((size_t)B * kC * 2304 * 4);
  float* asum  = (float*)alloc((size_t)B * 2304 * 4);
  bf16*  t_bf  = (bf16*)alloc((size_t)B * 2304 * 256 * 2);
  bf16*  qkvbf = (bf16*)alloc((size_t)B * 2304 * 768 * 2);
  bf16*  ostg  = (bf16*)alloc((size_t)B * 2304 * 256 * 2);
  float* xcat  = (float*)alloc((size_t)B * kNq * 256 * 4);
  bf16*  xcatb = (bf16*)alloc((size_t)B * kNq * 256 * 2);
  bf16*  Kbuf  = (bf16*)alloc((size_t)B * 32 * 1024 * 2);
  bf16*  Vbuf  = (bf16*)alloc((size_t)B * 32 * 1024 * 2);
  float* yf    = (float*)alloc((size_t)B * kNq * 256 * 4);
  float* x1f   = (float*)alloc((size_t)B * kNq * 256 * 4);
  bf16*  x1b   = (bf16*)alloc((size_t)B * kNq * 256 * 2);
  // Aliases (lifetimes disjoint): Q over xconv, O/H over xnorm, y2 over yf.
  bf16* Qbuf = (bf16*)xconv;   // B*3060*1024*2 = 12.5MB <= 18.9MB
  bf16* Obuf = (bf16*)xnorm;   // cross-attn output, then MLP hidden
  bf16* Hbuf = Obuf;
  float* y2 = yf;

  auto cvt = [&](const float* src, bf16* dst, int n) {
    k_f32_to_bf16<<<(n + 255) / 256, 256, 0, stream>>>(src, dst, n);
  };
  cvt(qkv_w, qkvw_bf, 4 * 768 * 256);
  cvt(proj_w, projw_bf, 4 * 256 * 256);
  cvt(wq, wq_bf, 1024 * 256);
  cvt(wk, wk_bf, 1024 * 256);
  cvt(wv, wv_bf, 1024 * 256);
  cvt(wo, wo_bf, 256 * 1024);
  cvt(m1_w, m1_bf, 1024 * 256);
  cvt(m2_w, m2_bf, 256 * 1024);
  cvt(lang, lang_bf, B * 32 * 256);

  const int hs[4] = {2, 4, 8, 16};
  const int offtok[4] = {0, 2304, 2880, 3024};
  const int total_pix = B * kC * kHW * kHW;

  for (int s = 0; s < 4; ++s) {
    int h = hs[s];
    int hh = kHW / h;
    int N = hh * hh;
    k_dwconv<<<(total_pix + 255) / 256, 256, 0, stream>>>(
        sp[s], pose_w + (size_t)s * 256 * 9, pose_b + s * 256, xconv, B);
    k_ln2d<<<B * kHW * kHW, 256, 0, stream>>>(xconv, pn_g + s * 256, pn_b + s * 256, xnorm, B);
    k_pool<<<(B * kC * N + 255) / 256, 256, 0, stream>>>(xnorm, stok, B, h);
    k_zero<<<(B * kC * N + 255) / 256, 256, 0, stream>>>(stacc, B * kC * N);
    k_zero<<<(B * N + 255) / 256, 256, 0, stream>>>(asum, B * N);
    k_affinity<<<B * N, 256, 0, stream>>>(xnorm, stok, stacc, asum, B, h);
    k_tokens<<<(B * N * kC + 255) / 256, 256, 0, stream>>>(stacc, asum, t_bf, B, h);
    // qkv = t @ qkv_w[s]^T + qkv_b[s]   (N x 768)
    {
      int tiles = ((N + 15) / 16) * (768 / 16);
      k_gemm_bt<<<dim3((tiles + 3) / 4, B), 128, 0, stream>>>(
          t_bf, 256, (long long)N * 256, qkvw_bf + (size_t)s * 768 * 256,
          qkv_b + s * 768, nullptr, 0, 0, qkvbf, 768, (long long)N * 768,
          nullptr, 0, 0, N, 768, 256, 0);
    }
    // MHSA over super tokens
    {
      int Npad = ((N + 31) / 32) * 32;
      size_t shm = (size_t)16 * Npad * sizeof(float);
      k_stage_attn<<<dim3((N + 15) / 16, kHeads, B), 32, shm, stream>>>(
          qkvbf, (long long)N * 768, N, Npad, ostg, (long long)N * 256,
          0.17677669529663687f /* 1/sqrt(32) */);
    }
    // proj into the concatenated token matrix (f32 + bf16 shadow)
    {
      int tiles = ((N + 15) / 16) * (256 / 16);
      k_gemm_bt<<<dim3((tiles + 3) / 4, B), 128, 0, stream>>>(
          ostg, 256, (long long)N * 256, projw_bf + (size_t)s * 256 * 256,
          proj_b + s * 256, xcat + (size_t)offtok[s] * 256, 256, (long long)kNq * 256,
          xcatb + (size_t)offtok[s] * 256, 256, (long long)kNq * 256,
          nullptr, 0, 0, N, 256, 256, 0);
    }
  }

  // Cross attention with language tokens
  {
    int tilesQ = ((kNq + 15) / 16) * (1024 / 16);
    k_gemm_bt<<<dim3((tilesQ + 3) / 4, B), 128, 0, stream>>>(
        xcatb, 256, (long long)kNq * 256, wq_bf, bq, nullptr, 0, 0,
        Qbuf, 1024, (long long)kNq * 1024, nullptr, 0, 0, kNq, 1024, 256, 0);
    int tilesKV = (32 / 16) * (1024 / 16);
    k_gemm_bt<<<dim3((tilesKV + 3) / 4, B), 128, 0, stream>>>(
        lang_bf, 256, (long long)32 * 256, wk_bf, bk, nullptr, 0, 0,
        Kbuf, 1024, (long long)32 * 1024, nullptr, 0, 0, 32, 1024, 256, 0);
    k_gemm_bt<<<dim3((tilesKV + 3) / 4, B), 128, 0, stream>>>(
        lang_bf, 256, (long long)32 * 256, wv_bf, bv, nullptr, 0, 0,
        Vbuf, 1024, (long long)32 * 1024, nullptr, 0, 0, 32, 1024, 256, 0);
    k_cross_attn<<<dim3((kNq + 15) / 16, kHeads, B), 32, 0, stream>>>(
        Qbuf, Kbuf, Vbuf, lmask, Obuf, kNq, 32);
    // y = xcat + O @ wo^T + bo
    int tilesO = ((kNq + 15) / 16) * (256 / 16);
    k_gemm_bt<<<dim3((tilesO + 3) / 4, B), 128, 0, stream>>>(
        Obuf, 1024, (long long)kNq * 1024, wo_bf, bo, yf, 256, (long long)kNq * 256,
        nullptr, 0, 0, xcat, 256, (long long)kNq * 256, kNq, 256, 1024, 0);
  }
  // x1 = LN(y)
  k_ln<<<B * kNq, 256, 0, stream>>>(yf, n1_g, n1_b, x1f, x1b, B * kNq);
  // MLP: H = relu(x1 @ m1^T + b1); y2 = x1 + H @ m2^T + b2
  {
    int tiles1 = ((kNq + 15) / 16) * (1024 / 16);
    k_gemm_bt<<<dim3((tiles1 + 3) / 4, B), 128, 0, stream>>>(
        x1b, 256, (long long)kNq * 256, m1_bf, m1_b, nullptr, 0, 0,
        Hbuf, 1024, (long long)kNq * 1024, nullptr, 0, 0, kNq, 1024, 256, 1);
    int tiles2 = ((kNq + 15) / 16) * (256 / 16);
    k_gemm_bt<<<dim3((tiles2 + 3) / 4, B), 128, 0, stream>>>(
        Hbuf, 1024, (long long)kNq * 1024, m2_bf, m2_b, y2, 256, (long long)kNq * 256,
        nullptr, 0, 0, x1f, 256, (long long)kNq * 256, kNq, 256, 1024, 0);
  }
  // out = LN(y2)
  k_ln<<<B * kNq, 256, 0, stream>>>(y2, n2_g, n2_b, (float*)d_out, nullptr, B * kNq);
}
